// Model_1_12936441496279
// MI455X (gfx1250) — matrix operand order repaired and checked by reading, not compiled, not run on hardware
//
#include <hip/hip_runtime.h>

// ---------------------------------------------------------------------------
// Problem constants (from reference)
// ---------------------------------------------------------------------------
#define BGR   32            // graphs per batch
#define N0    2048          // nodes per graph at level 0
#define NTOT  (BGR * N0)    // 65536 total nodes
#define FDIM  128           // feature dim
#define NRREL 4             // relations
#define NBAS  2             // bases
#define ETOT  (BGR * 8192)  // 262144 edges
#define GCOLS 384           // fused GEMM output cols: basis0|basis1|root

typedef __attribute__((ext_vector_type(16))) __bf16 v16bf;
typedef __attribute__((ext_vector_type(8)))  float  v8f;

__device__ __forceinline__ float leaky(float v) { return v >= 0.0f ? v : 0.01f * v; }

// ---------------------------------------------------------------------------
// Utility fills
// ---------------------------------------------------------------------------
__global__ void fill_f32(float* p, float v, size_t n) {
    size_t i = (size_t)blockIdx.x * blockDim.x + threadIdx.x;
    if (i < n) p[i] = v;
}
__global__ void fill_i32(int* p, int v, size_t n) {
    size_t i = (size_t)blockIdx.x * blockDim.x + threadIdx.x;
    if (i < n) p[i] = v;
}

// ---------------------------------------------------------------------------
// BatchNorm (training-mode batch stats over all nodes)
// ---------------------------------------------------------------------------
__global__ void bn_reduce(const float* __restrict__ x, float* __restrict__ stat, int nrows) {
    __shared__ float s1[256], s2[256];
    int f = blockIdx.x, t = threadIdx.x;
    float a = 0.0f, b = 0.0f;
    for (int i = t; i < nrows; i += 256) {
        float v = x[(size_t)i * FDIM + f];
        a += v; b += v * v;
    }
    s1[t] = a; s2[t] = b; __syncthreads();
    for (int s = 128; s > 0; s >>= 1) {
        if (t < s) { s1[t] += s1[t + s]; s2[t] += s2[t + s]; }
        __syncthreads();
    }
    if (t == 0) {
        float mu = s1[0] / nrows;
        stat[f] = mu;
        stat[FDIM + f] = s2[0] / nrows - mu * mu;
    }
}
__global__ void bn_apply(const float* __restrict__ x, const float* __restrict__ stat,
                         const float* __restrict__ g, const float* __restrict__ be,
                         float* __restrict__ xn, size_t n) {
    size_t i = (size_t)blockIdx.x * blockDim.x + threadIdx.x;
    if (i >= n) return;
    int f = (int)(i & (FDIM - 1));
    xn[i] = (x[i] - stat[f]) * rsqrtf(stat[FDIM + f] + 1e-5f) * g[f] + be[f];
}

// ---------------------------------------------------------------------------
// Pre-convert per-layer weights into transposed bf16 [GCOLS][128]:
//   col c in [0,128): basis[0][k][c]; [128,256): basis[1][k][c-128]; [256,384): root[k][c-256]
// Transposed layout makes K-pairs contiguous for the B-fragment loads.
// ---------------------------------------------------------------------------
__global__ void conv_weights_bf16(const float* __restrict__ basis, const float* __restrict__ root,
                                  __bf16* __restrict__ wt) {
    int c = blockIdx.x;        // output column  [0, 384)
    int k = threadIdx.x;       // input feature  [0, 128)
    float v;
    if (c < 128)       v = basis[(size_t)k * 128 + c];
    else if (c < 256)  v = basis[(size_t)128 * 128 + (size_t)k * 128 + (c - 128)];
    else               v = root[(size_t)k * 128 + (c - 256)];
    wt[(size_t)c * 128 + k] = (__bf16)v;
}

// ---------------------------------------------------------------------------
// WMMA GEMM: out[nrows, 384] = x[nrows, 128] @ Wcat[128, 384]  (bf16 in, f32 acc)
// Block = 128 threads = 4 waves. Each wave owns a 16-row slab and 128 output
// columns (8 C tiles): the A fragment is built once per K-step (fp32 load +
// bf16 cvt) and reused across 8 back-to-back v_wmma issues, cutting A traffic
// and conversion work 8x vs one-tile-per-wave. K=128 as four K=32 WMMA steps
// -> 32 static v_wmma per kernel.
// A layout (16-bit A 16x32): lane group g (lane>=16), row m = lane&15,
//   VGPR j holds K = (j<4?0:16) + g*8 + 2*(j&3) .. +1
// B layout (16-bit B 32x16): col n = lane&15, VGPR j holds K = (j<4?0:16) + g*8 + 2*(j&3) .. +1
// C layout: VGPR r -> row m = r + g*8, col = lane&15
// ---------------------------------------------------------------------------
#define CTILES 8   // 8 x 16 = 128 output columns per wave
__global__ void __launch_bounds__(128)
rgcn_gemm_wmma(const float* __restrict__ x, const __bf16* __restrict__ wt,
               float* __restrict__ out, int nrows) {
    const int lane = threadIdx.x & 31;
    const int wave = threadIdx.x >> 5;
    const int row0 = (blockIdx.x * 4 + wave) * 16;
    const int col0 = blockIdx.y * (CTILES * 16);
    if (row0 >= nrows) return;
    const int g = lane >> 4;
    const int m = lane & 15;
    v8f acc[CTILES] = {};
#pragma unroll
    for (int kk = 0; kk < 4; ++kk) {
        // Build A fragment once per K-step
        v16bf a;
        const float* xrow = x + (size_t)(row0 + m) * FDIM + kk * 32;
#pragma unroll
        for (int j = 0; j < 8; ++j) {
            int kb = ((j < 4) ? 0 : 16) + g * 8 + 2 * (j & 3);
            float2 av = *(const float2*)(xrow + kb);
            a[2 * j]     = (__bf16)av.x;
            a[2 * j + 1] = (__bf16)av.y;
        }
        // Sweep 8 column tiles, reusing A
#pragma unroll
        for (int ct = 0; ct < CTILES; ++ct) {
            v16bf b;
            const __bf16* wrow = wt + (size_t)(col0 + ct * 16 + m) * FDIM + kk * 32;
#pragma unroll
            for (int j = 0; j < 8; ++j) {
                int kb = ((j < 4) ? 0 : 16) + g * 8 + 2 * (j & 3);
                b[2 * j]     = wrow[kb];
                b[2 * j + 1] = wrow[kb + 1];
            }
            acc[ct] = __builtin_amdgcn_wmma_f32_16x16x32_bf16(false, a, false, b,
                                                              (short)0, acc[ct], false, false);
        }
    }
#pragma unroll
    for (int ct = 0; ct < CTILES; ++ct) {
        float* orow = out + (size_t)row0 * GCOLS + col0 + ct * 16 + m;
#pragma unroll
        for (int r = 0; r < 8; ++r)
            orow[(size_t)(r + g * 8) * GCOLS] = acc[ct][r];
    }
}

// ---------------------------------------------------------------------------
// Edge kernels: per-(dst,rel) valid-edge counts, then normalized message scatter
// ---------------------------------------------------------------------------
__global__ void edge_count(const int* __restrict__ dst, const int* __restrict__ et,
                           const int* __restrict__ em, float* __restrict__ cnt, int ne) {
    int e = blockIdx.x * blockDim.x + threadIdx.x;
    if (e >= ne || !em[e]) return;
    atomicAdd(&cnt[(size_t)dst[e] * NRREL + et[e]], 1.0f);
}

__global__ void edge_msg(const float* __restrict__ gout, const int* __restrict__ src,
                         const int* __restrict__ dst, const int* __restrict__ et,
                         const int* __restrict__ em, const float* __restrict__ comp,
                         const float* __restrict__ cnt, float* __restrict__ agg, int ne) {
    int e = blockIdx.x;
    int f = threadIdx.x;          // 128 threads = feature dim
    if (e >= ne || !em[e]) return;
    int s = src[e], d = dst[e], t = et[e];
    float c0 = comp[t * NBAS + 0], c1 = comp[t * NBAS + 1];
    float msg = c0 * gout[(size_t)s * GCOLS + f] + c1 * gout[(size_t)s * GCOLS + 128 + f];
    float norm = 1.0f / fmaxf(cnt[(size_t)d * NRREL + t], 1.0f);
    atomicAdd(&agg[(size_t)d * FDIM + f], msg * norm);
}

// xt = leaky(agg + root-part-of-gout + bias)
__global__ void finish_conv(const float* __restrict__ agg, const float* __restrict__ gout,
                            const float* __restrict__ bias, float* __restrict__ xt, size_t total) {
    size_t i = (size_t)blockIdx.x * blockDim.x + threadIdx.x;
    if (i >= total) return;
    size_t node = i >> 7;
    int f = (int)(i & 127);
    xt[i] = leaky(agg[i] + gout[node * GCOLS + 256 + f] + bias[f]);
}

// ---------------------------------------------------------------------------
// TopK pooling (ratio 0.5): per-graph bitonic sort of scores in LDS.
// n is a power of two (2048/1024/512). Writes pooled x and inverse index map.
// ---------------------------------------------------------------------------
__global__ void __launch_bounds__(1024)
topk_pool(const float* __restrict__ xt, const float* __restrict__ w,
          float* __restrict__ xout, int* __restrict__ inv, int n) {
    __shared__ float ssc[2048];
    __shared__ int   sidx[2048];
    __shared__ float swn;
    int b = blockIdx.x, t = threadIdx.x;
    const float* xg = xt + (size_t)b * n * FDIM;
    if (t == 0) {
        float s = 0.0f;
        for (int i = 0; i < FDIM; ++i) s += w[i] * w[i];
        swn = rsqrtf(fmaxf(s, 1e-20f));
    }
    __syncthreads();
    for (int i = t; i < n; i += blockDim.x) {
        float s = 0.0f;
        const float* xr = xg + (size_t)i * FDIM;
        for (int f = 0; f < FDIM; ++f) s += xr[f] * w[f];
        ssc[i] = s * swn;
        sidx[i] = i;
    }
    __syncthreads();
    // bitonic sort, descending
    for (int ksz = 2; ksz <= n; ksz <<= 1) {
        for (int jst = ksz >> 1; jst > 0; jst >>= 1) {
            for (int i = t; i < n; i += blockDim.x) {
                int ixj = i ^ jst;
                if (ixj > i) {
                    float a = ssc[i], c = ssc[ixj];
                    bool up = ((i & ksz) == 0);
                    bool sw = up ? (a < c) : (a > c);
                    if (sw) {
                        ssc[i] = c; ssc[ixj] = a;
                        int ti = sidx[i]; sidx[i] = sidx[ixj]; sidx[ixj] = ti;
                    }
                }
            }
            __syncthreads();
        }
    }
    int k = n >> 1;
    for (int idx = t; idx < k * FDIM; idx += blockDim.x) {
        int j = idx >> 7, f = idx & 127;
        xout[((size_t)b * k + j) * FDIM + f] =
            xg[(size_t)sidx[j] * FDIM + f] * tanhf(ssc[j]);
    }
    for (int j = t; j < k; j += blockDim.x)
        inv[b * n + sidx[j]] = b * k + j;
}

__global__ void remap_edges(const int* __restrict__ inv,
                            const int* si, const int* di, const int* ti, const int* mi,
                            int* so, int* dd, int* to, int* mo, int ne) {
    int e = blockIdx.x * blockDim.x + threadIdx.x;
    if (e >= ne) return;
    int s2 = inv[si[e]], d2 = inv[di[e]];
    mo[e] = (mi[e] && s2 >= 0 && d2 >= 0) ? 1 : 0;
    so[e] = s2 > 0 ? s2 : 0;
    dd[e] = d2 > 0 ? d2 : 0;
    to[e] = ti[e];
}

// ---------------------------------------------------------------------------
// Readouts (each accumulates into racc across the 3 pooling levels)
// ---------------------------------------------------------------------------
__global__ void readout_maxmean(const float* __restrict__ xp, float* __restrict__ racc, int n) {
    int b = blockIdx.x, f = threadIdx.x;  // 128 threads
    const float* xg = xp + (size_t)b * n * FDIM;
    float mx = -3.4e38f, sm = 0.0f;
    for (int i = 0; i < n; ++i) {
        float v = xg[(size_t)i * FDIM + f];
        mx = fmaxf(mx, v); sm += v;
    }
    racc[b * 256 + f]       += mx;
    racc[b * 256 + 128 + f] += sm / (float)n;
}

__global__ void readout_set2set(const float* __restrict__ xp,
                                const float* __restrict__ wih, const float* __restrict__ whh,
                                const float* __restrict__ bih, const float* __restrict__ bhh,
                                float* __restrict__ racc, int n) {
    __shared__ float h[FDIM], c[FDIM], q[2 * FDIM];
    __shared__ float logit[2048];
    __shared__ float red[FDIM];
    int b = blockIdx.x, t = threadIdx.x;  // 128 threads
    const float* xg = xp + (size_t)b * n * FDIM;
    h[t] = 0.0f; c[t] = 0.0f; q[t] = 0.0f; q[t + FDIM] = 0.0f;
    __syncthreads();
    for (int step = 0; step < 3; ++step) {
        float gate[4];
        for (int gi = 0; gi < 4; ++gi) {
            int row = gi * FDIM + t;
            float s = bih[row] + bhh[row];
            const float* wr = wih + (size_t)row * (2 * FDIM);
            for (int i = 0; i < 2 * FDIM; ++i) s += q[i] * wr[i];
            const float* wr2 = whh + (size_t)row * FDIM;
            for (int i = 0; i < FDIM; ++i) s += h[i] * wr2[i];
            gate[gi] = s;
        }
        __syncthreads();
        float ig = 1.0f / (1.0f + expf(-gate[0]));
        float fg = 1.0f / (1.0f + expf(-gate[1]));
        float gg = tanhf(gate[2]);
        float og = 1.0f / (1.0f + expf(-gate[3]));
        float cn = fg * c[t] + ig * gg;
        c[t] = cn; h[t] = og * tanhf(cn);
        __syncthreads();
        for (int i = t; i < n; i += FDIM) {
            float s = 0.0f;
            const float* xr = xg + (size_t)i * FDIM;
            for (int f = 0; f < FDIM; ++f) s += xr[f] * h[f];
            logit[i] = s;
        }
        __syncthreads();
        float mx = -3.4e38f;
        for (int i = t; i < n; i += FDIM) mx = fmaxf(mx, logit[i]);
        red[t] = mx; __syncthreads();
        for (int s = 64; s > 0; s >>= 1) { if (t < s) red[t] = fmaxf(red[t], red[t + s]); __syncthreads(); }
        float gmax = red[0]; __syncthreads();
        float sm = 0.0f;
        for (int i = t; i < n; i += FDIM) { float e = expf(logit[i] - gmax); logit[i] = e; sm += e; }
        red[t] = sm; __syncthreads();
        for (int s = 64; s > 0; s >>= 1) { if (t < s) red[t] += red[t + s]; __syncthreads(); }
        float tot = red[0]; __syncthreads();
        float r = 0.0f;
        for (int i = 0; i < n; ++i) r += logit[i] * xg[(size_t)i * FDIM + t];
        r /= tot;
        q[t] = h[t]; q[t + FDIM] = r;
        __syncthreads();
    }
    racc[b * 256 + t]       += q[t];
    racc[b * 256 + 128 + t] += q[t + FDIM];
}

__global__ void readout_sortpool(const float* __restrict__ xp, float* __restrict__ racc, int n) {
    __shared__ float sv[FDIM];
    __shared__ int   si[FDIM];
    int b = blockIdx.x, t = threadIdx.x;  // 128 threads
    const float* xg = xp + (size_t)b * n * FDIM;
    float best = -3.4e38f; int bi = 0;
    for (int i = t; i < n; i += FDIM) {
        float v = xg[(size_t)i * FDIM + 127];
        if (v > best) { best = v; bi = i; }
    }
    sv[t] = best; si[t] = bi; __syncthreads();
    for (int s = 64; s > 0; s >>= 1) {
        if (t < s && sv[t + s] > sv[t]) { sv[t] = sv[t + s]; si[t] = si[t + s]; }
        __syncthreads();
    }
    racc[b * FDIM + t] += xg[(size_t)si[0] * FDIM + t];
}

// ---------------------------------------------------------------------------
// Final MLP head + log_softmax, one block per graph
// ---------------------------------------------------------------------------
__global__ void mlp_head(const float* __restrict__ r1, const float* __restrict__ r2,
                         const float* __restrict__ r3,
                         const float* __restrict__ w1, const float* __restrict__ b1,
                         const float* __restrict__ w2, const float* __restrict__ b2,
                         const float* __restrict__ w3, const float* __restrict__ b3,
                         float* __restrict__ out) {
    __shared__ float hc[640], h1[128], h2[64], lg[2];
    int b = blockIdx.x, t = threadIdx.x;  // 128 threads
    hc[t]       = r1[b * 256 + t];
    hc[128 + t] = r1[b * 256 + 128 + t];
    hc[256 + t] = r2[b * 256 + t];
    hc[384 + t] = r2[b * 256 + 128 + t];
    hc[512 + t] = r3[b * 128 + t];
    __syncthreads();
    float s = b1[t];
    for (int i = 0; i < 640; ++i) s += hc[i] * w1[(size_t)i * 128 + t];
    h1[t] = leaky(s);
    __syncthreads();
    if (t < 64) {
        float s2 = b2[t];
        for (int i = 0; i < 128; ++i) s2 += h1[i] * w2[(size_t)i * 64 + t];
        h2[t] = leaky(s2);
    }
    __syncthreads();
    if (t < 2) {
        float s3 = b3[t];
        for (int i = 0; i < 64; ++i) s3 += h2[i] * w3[(size_t)i * 2 + t];
        lg[t] = s3;
    }
    __syncthreads();
    if (t < 2) {
        float m = fmaxf(lg[0], lg[1]);
        float lse = m + logf(expf(lg[0] - m) + expf(lg[1] - m));
        out[b * 2 + t] = lg[t] - lse;
    }
}

// ---------------------------------------------------------------------------
// Host orchestration
// ---------------------------------------------------------------------------
__global__ void edge_init(const int* __restrict__ ei, const int* __restrict__ etin,
                          int* s, int* d, int* t, int* m, int ne) {
    int e = blockIdx.x * blockDim.x + threadIdx.x;
    if (e >= ne) return;
    s[e] = ei[e];
    d[e] = ei[ne + e];
    t[e] = etin[e];
    m[e] = 1;
}

extern "C" void kernel_launch(void* const* d_in, const int* in_sizes, int n_in,
                              void* d_out, int out_size, void* d_ws, size_t ws_size,
                              hipStream_t stream) {
    (void)in_sizes; (void)n_in; (void)out_size; (void)ws_size;
    // Inputs: x, edge_index, edge_type, batch, then params flattened in JAX
    // pytree (alphabetical-key) order.
    const float* x   = (const float*)d_in[0];
    const int*   ei  = (const int*)d_in[1];
    const int*   ety = (const int*)d_in[2];
    const float* bn_b = (const float*)d_in[4];
    const float* bn_g = (const float*)d_in[5];
    const float* basisP[9]; const float* biasP[9]; const float* compP[9];
    const float* rootP[9];  const float* poolP[9];
    for (int i = 0; i < 9; ++i) {
        basisP[i] = (const float*)d_in[6 + 4 * i];
        biasP[i]  = (const float*)d_in[7 + 4 * i];
        compP[i]  = (const float*)d_in[8 + 4 * i];
        rootP[i]  = (const float*)d_in[9 + 4 * i];
        poolP[i]  = (const float*)d_in[52 + i];
    }
    const float* lin1_b = (const float*)d_in[42]; const float* lin1_w = (const float*)d_in[43];
    const float* lin2_b = (const float*)d_in[44]; const float* lin2_w = (const float*)d_in[45];
    const float* lin3_b = (const float*)d_in[46]; const float* lin3_w = (const float*)d_in[47];
    const float* bhh = (const float*)d_in[48]; const float* bih = (const float*)d_in[49];
    const float* whh = (const float*)d_in[50]; const float* wih = (const float*)d_in[51];

    // Workspace carve-out
    char* wsp = (char*)d_ws;
    size_t off = 0;
    auto take = [&](size_t bytes) -> char* {
        char* p = wsp + off;
        off = (off + bytes + 255) & ~(size_t)255;
        return p;
    };
    float* xn   = (float*)take((size_t)NTOT * FDIM * 4);
    float* xA   = (float*)take((size_t)NTOT * FDIM * 4);   // current node feats / pooled output
    float* xB   = (float*)take((size_t)NTOT * FDIM * 4);   // post-conv feats
    float* gout = (float*)take((size_t)NTOT * GCOLS * 4);  // WMMA GEMM output
    float* agg  = (float*)take((size_t)NTOT * FDIM * 4);
    float* cnt  = (float*)take((size_t)NTOT * NRREL * 4);
    int*   inv  = (int*)take((size_t)NTOT * 4);
    int *es[2], *ed[2], *et2[2], *em2[2];
    for (int i = 0; i < 2; ++i) {
        es[i]  = (int*)take((size_t)ETOT * 4);
        ed[i]  = (int*)take((size_t)ETOT * 4);
        et2[i] = (int*)take((size_t)ETOT * 4);
        em2[i] = (int*)take((size_t)ETOT * 4);
    }
    __bf16* wbf  = (__bf16*)take((size_t)9 * GCOLS * FDIM * 2);
    float* stat  = (float*)take(256 * 4);
    float* racc1 = (float*)take((size_t)BGR * 256 * 4);
    float* racc2 = (float*)take((size_t)BGR * 256 * 4);
    float* racc3 = (float*)take((size_t)BGR * 128 * 4);

    const int TB = 256;
    auto blocks = [](size_t n, int tb) { return (unsigned)((n + tb - 1) / tb); };

    // Zero accumulators (must be re-zeroed every call; harness does not restore ws)
    fill_f32<<<blocks(BGR * 256, TB), TB, 0, stream>>>(racc1, 0.0f, BGR * 256);
    fill_f32<<<blocks(BGR * 256, TB), TB, 0, stream>>>(racc2, 0.0f, BGR * 256);
    fill_f32<<<blocks(BGR * 128, TB), TB, 0, stream>>>(racc3, 0.0f, BGR * 128);

    // BatchNorm
    bn_reduce<<<FDIM, 256, 0, stream>>>(x, stat, NTOT);
    bn_apply<<<blocks((size_t)NTOT * FDIM, TB), TB, 0, stream>>>(x, stat, bn_g, bn_b, xn,
                                                                 (size_t)NTOT * FDIM);

    // Pre-convert all 9 fused weight matrices to transposed bf16
    for (int l = 0; l < 9; ++l)
        conv_weights_bf16<<<GCOLS, FDIM, 0, stream>>>(basisP[l], rootP[l],
                                                      wbf + (size_t)l * GCOLS * FDIM);

    for (int br = 0; br < 3; ++br) {
        edge_init<<<blocks(ETOT, TB), TB, 0, stream>>>(ei, ety, es[0], ed[0], et2[0], em2[0], ETOT);
        hipMemcpyAsync(xA, xn, (size_t)NTOT * FDIM * 4, hipMemcpyDeviceToDevice, stream);
        int n = N0, cur = 0;
        for (int lv = 0; lv < 3; ++lv) {
            int layer = br * 3 + lv;
            int nrows = BGR * n;
            fill_f32<<<blocks((size_t)nrows * NRREL, TB), TB, 0, stream>>>(cnt, 0.0f,
                                                                           (size_t)nrows * NRREL);
            fill_f32<<<blocks((size_t)nrows * FDIM, TB), TB, 0, stream>>>(agg, 0.0f,
                                                                          (size_t)nrows * FDIM);
            // Fused basis/root transform via WMMA: 4 row-tiles per block, 3 col slabs
            rgcn_gemm_wmma<<<dim3(nrows / 64, GCOLS / (CTILES * 16)), 128, 0, stream>>>(
                xA, wbf + (size_t)layer * GCOLS * FDIM, gout, nrows);
            // Relation-mean message passing
            edge_count<<<blocks(ETOT, TB), TB, 0, stream>>>(ed[cur], et2[cur], em2[cur], cnt, ETOT);
            edge_msg<<<ETOT, FDIM, 0, stream>>>(gout, es[cur], ed[cur], et2[cur], em2[cur],
                                                compP[layer], cnt, agg, ETOT);
            finish_conv<<<blocks((size_t)nrows * FDIM, TB), TB, 0, stream>>>(
                agg, gout, biasP[layer], xB, (size_t)nrows * FDIM);
            // TopK pooling
            fill_i32<<<blocks((size_t)BGR * n, TB), TB, 0, stream>>>(inv, -1, (size_t)BGR * n);
            topk_pool<<<BGR, 1024, 0, stream>>>(xB, poolP[layer], xA, inv, n);
            remap_edges<<<blocks(ETOT, TB), TB, 0, stream>>>(
                inv, es[cur], ed[cur], et2[cur], em2[cur],
                es[cur ^ 1], ed[cur ^ 1], et2[cur ^ 1], em2[cur ^ 1], ETOT);
            cur ^= 1;
            n >>= 1;
            // Readout on pooled nodes
            if (br == 0)
                readout_maxmean<<<BGR, FDIM, 0, stream>>>(xA, racc1, n);
            else if (br == 1)
                readout_set2set<<<BGR, FDIM, 0, stream>>>(xA, wih, whh, bih, bhh, racc2, n);
            else
                readout_sortpool<<<BGR, FDIM, 0, stream>>>(xA, racc3, n);
        }
    }

    mlp_head<<<BGR, FDIM, 0, stream>>>(racc1, racc2, racc3, lin1_w, lin1_b,
                                       lin2_w, lin2_b, lin3_w, lin3_b, (float*)d_out);
}
